// GeneSetPlaceholderAggregator_4140348473467
// MI455X (gfx1250) — matrix operand order repaired and checked by reading, not compiled, not run on hardware
//
#include <hip/hip_runtime.h>

// Problem constants (from reference): B=64, G=20000, D=16, S=128
#define G_      20000
#define B_      64
#define S_      128
#define D_      16
#define KS_     20                // K-split factor (round-robin over chunks)
#define NCHUNK_ 625               // G / 32
#define OUTSZ_  (B_ * S_ * D_)    // 131072 floats
#define BPW_    4                 // batches per wave

typedef __attribute__((ext_vector_type(16))) __bf16 v16bf;
typedef __attribute__((ext_vector_type(8)))  float  v8f;
typedef __attribute__((ext_vector_type(2)))  float  f32x2;
typedef __attribute__((ext_vector_type(2)))  __bf16 bf16x2;
typedef unsigned int u32;

// Reinterpret 8 packed bf16 pairs as a WMMA operand. Pair i = VGPR i of the
// operand, i.e. elements (2i, 2i+1) — exactly the in-register layout.
union V16 {
    v16bf bf;
    u32   u[8];
};

// One packed hi/lo bf16 split: f = hi + lo (to ~16-bit mantissa accuracy).
// Lowers to: v_cvt_pk_bf16_f32 (hi), lshl/and up-convert, v_pk_add_f32, cvt (lo).
__device__ inline void split2(f32x2 f, u32& hi, u32& lo) {
    bf16x2 h  = __builtin_convertvector(f, bf16x2);
    f32x2  hf = __builtin_convertvector(h, f32x2);
    bf16x2 l  = __builtin_convertvector(f - hf, bf16x2);
    hi = __builtin_bit_cast(u32, h);
    lo = __builtin_bit_cast(u32, l);
}

__device__ inline void split16(const f32x2* f, v16bf& hi, v16bf& lo) {
    V16 h, l;
#pragma unroll
    for (int i = 0; i < 8; ++i) split2(f[i], h.u[i], l.u[i]);
    hi = h.bf;
    lo = l.bf;
}

// Load 16 contiguous floats (64B) as 8 f32x2 pairs.
__device__ inline void ld16(const float* __restrict__ p, f32x2* d) {
    const float4* q = reinterpret_cast<const float4*>(p);
    float4 a = q[0], b = q[1], c = q[2], e = q[3];
    d[0] = f32x2{a.x, a.y}; d[1] = f32x2{a.z, a.w};
    d[2] = f32x2{b.x, b.y}; d[3] = f32x2{b.z, b.w};
    d[4] = f32x2{c.x, c.y}; d[5] = f32x2{c.z, c.w};
    d[6] = f32x2{e.x, e.y}; d[7] = f32x2{e.z, e.w};
}

// A-operand for one lane: 8 floats at (row+off) and 8 at (row+off+16),
// matching the 16-bit A 16x32 VGPR layout (two K-halves per lane).
__device__ inline void ldA(const float* __restrict__ row, int off, f32x2* d) {
    const float4* q0 = reinterpret_cast<const float4*>(row + off);
    const float4* q1 = reinterpret_cast<const float4*>(row + off + 16);
    float4 a = q0[0], b = q0[1], c = q1[0], e = q1[1];
    d[0] = f32x2{a.x, a.y}; d[1] = f32x2{a.z, a.w};
    d[2] = f32x2{b.x, b.y}; d[3] = f32x2{b.z, b.w};
    d[4] = f32x2{c.x, c.y}; d[5] = f32x2{c.z, c.w};
    d[6] = f32x2{e.x, e.y}; d[7] = f32x2{e.z, e.w};
}

__device__ inline v8f wmma3(v16bf ah, v16bf al, v16bf bh, v16bf bl, v8f acc) {
    // (ah+al)*(bh+bl) ~= ah*bh + ah*bl + al*bh   (al*bl below f32 ulp)
    acc = __builtin_amdgcn_wmma_f32_16x16x32_bf16(false, ah, false, bh, (short)0, acc, false, false);
    acc = __builtin_amdgcn_wmma_f32_16x16x32_bf16(false, ah, false, bl, (short)0, acc, false, false);
    acc = __builtin_amdgcn_wmma_f32_16x16x32_bf16(false, al, false, bh, (short)0, acc, false, false);
    return acc;
}

// Each wave: 2 s-tiles (16 rows each) x 4 batches x 1/KS of K.
// Grid: 160 blocks x 256 threads = 1280 waves = 4 s-groups * 16 b-groups * 20 k-slices.
template <bool ATOMIC>
__global__ __launch_bounds__(256) void geneset_wmma_kernel(
    const float* __restrict__ x,   // [B, G, D] f32
    const float* __restrict__ W,   // [S, G]    f32
    float* __restrict__ out)       // ATOMIC: [B,S,D] ; else partials [KS][B,S,D]
{
    const int lane = threadIdx.x & 31;
    const int wav  = threadIdx.x >> 5;
    const int gw   = blockIdx.x * 8 + wav;      // 0..1279
    const int kk   = gw % KS_;
    const int t    = gw / KS_;                  // 0..63
    const int bg   = t & 15;                    // 16 batch groups
    const int sg   = t >> 4;                    // 4 s-tile groups
    const int b0   = bg * BPW_;                 // batches b0 .. b0+3
    const int s0   = sg << 5;                   // s-tiles [s0, s0+32)

    // A layout (16-bit 16x32): lanes 0-15 carry K-halves {0..7,16..23},
    // lanes 16-31 carry {8..15,24..31}; M = lane & 15.
    const int kb   = (lane & 16) ? 8 : 0;
    const int mrow = lane & 15;

    const float* Wr0 = W + (size_t)(s0 + mrow) * G_;
    const float* Wr1 = W + (size_t)(s0 + 16 + mrow) * G_;
    const float* X0  = x + (size_t)b0 * G_ * D_;

    v8f acc0[BPW_] = {}, acc1[BPW_] = {};   // [tile][batch]

    for (int c = kk; c < NCHUNK_; c += KS_) {
        const int g = c << 5;   // gene base of this K=32 chunk

        // Prefetch this wave's next chunk (speculative; OOB is dropped).
        __builtin_prefetch(Wr0 + g + KS_ * 32 + kb, 0, 1);
        __builtin_prefetch(X0 + (size_t)(g + KS_ * 32 + lane) * D_, 0, 1);

        // A tiles first — held live across all batches.
        f32x2 av[8];
        v16bf ah0, al0, ah1, al1;
        ldA(Wr0, g + kb, av);
        split16(av, ah0, al0);
        ldA(Wr1, g + kb, av);
        split16(av, ah1, al1);

        // B operand per batch: lane L holds N = d = L & 15, K = genes in A's table order.
#pragma unroll
        for (int bb = 0; bb < BPW_; ++bb) {
            f32x2 xr[8];
            v16bf bh, bl;
            {
                const float* p = X0 + ((size_t)bb * G_ + (g + kb)) * D_ + (lane & 15);
#pragma unroll
                for (int i = 0; i < 8; ++i) {
                    const int k = (i < 4) ? 2 * i : 16 + 2 * (i - 4);
                    xr[i] = f32x2{p[(size_t)k * D_], p[(size_t)(k + 1) * D_]};
                }
            }
            split16(xr, bh, bl);
            acc0[bb] = wmma3(ah0, al0, bh, bl, acc0[bb]);
            acc1[bb] = wmma3(ah1, al1, bh, bl, acc1[bb]);
        }
    }

    // C/D layout: VGPR i -> M = i (+8 for lanes 16-31), N = lane & 15.
    const int nlo   = lane & 15;
    const int shalf = (lane & 16) ? 8 : 0;
    float* o = ATOMIC ? out : out + (size_t)kk * OUTSZ_;

#pragma unroll
    for (int bb = 0; bb < BPW_; ++bb) {
        const size_t e0 = ((size_t)(b0 + bb) * S_ + s0 + shalf) * D_ + nlo;
        const size_t e1 = e0 + (size_t)16 * D_;   // second s-tile
#pragma unroll
        for (int i = 0; i < 8; ++i) {
            const size_t r = (size_t)i * D_;
            if (ATOMIC) {
                unsafeAtomicAdd(&o[e0 + r], acc0[bb][i]);
                unsafeAtomicAdd(&o[e1 + r], acc1[bb][i]);
            } else {
                o[e0 + r] = acc0[bb][i];
                o[e1 + r] = acc1[bb][i];
            }
        }
    }
}

// Deterministic K-slice reduction: out[i] = sum_k partial[k][i].
__global__ __launch_bounds__(256) void geneset_reduce_kernel(
    const float* __restrict__ part, float* __restrict__ out)
{
    const int i = blockIdx.x * blockDim.x + threadIdx.x;
    if (i < OUTSZ_) {
        float s = 0.f;
#pragma unroll
        for (int k = 0; k < KS_; ++k) s += part[(size_t)k * OUTSZ_ + i];
        out[i] = s;
    }
}

extern "C" void kernel_launch(void* const* d_in, const int* in_sizes, int n_in,
                              void* d_out, int out_size, void* d_ws, size_t ws_size,
                              hipStream_t stream) {
    const float* x = (const float*)d_in[0];   // [64, 20000, 16] f32
    const float* W = (const float*)d_in[1];   // [128, 20000]    f32
    float* out = (float*)d_out;               // [64, 128, 16]   f32

    const dim3 grid(160), block(256);
    const size_t need = (size_t)KS_ * OUTSZ_ * sizeof(float);  // ~10.5 MB

    if (ws_size >= need) {
        // Deterministic path: per-K-slice partials in workspace, then reduce.
        float* part = (float*)d_ws;
        geneset_wmma_kernel<false><<<grid, block, 0, stream>>>(x, W, part);
        geneset_reduce_kernel<<<(OUTSZ_ + 255) / 256, 256, 0, stream>>>(part, out);
    } else {
        // Fallback: zero output then atomically accumulate K slices.
        hipMemsetAsync(out, 0, (size_t)OUTSZ_ * sizeof(float), stream);
        geneset_wmma_kernel<true><<<grid, block, 0, stream>>>(x, W, out);
    }
}
